// ImprovedMultiHeadAttention_15522011808294
// MI455X (gfx1250) — matrix operand order repaired and checked by reading, not compiled, not run on hardware
//
#include <hip/hip_runtime.h>
#include <hip/hip_bf16.h>

// ---------------------------------------------------------------------------
// ImprovedMultiHeadAttention forward for MI455X (gfx1250), wave32 + WMMA.
//  Stage 1 (proj_kernel): Q = (x@Wq^T + bq + pos) * 1/8, K = x@Wk^T + bk + pos,
//                         V = key@Wv^T + bv  -> bf16; V stored transposed.
//  Stage 2 (attn_kernel): flash-style online-softmax attention, all matmuls
//                         via v_wmma_f32_16x16x32_bf16.
// ---------------------------------------------------------------------------

typedef __attribute__((ext_vector_type(16))) __bf16 v16bf;
typedef __attribute__((ext_vector_type(8)))  __bf16 v8bf;
typedef __attribute__((ext_vector_type(8)))  float  v8f;

static constexpr int D_MODEL = 1024;
static constexpr int HEADS   = 16;
static constexpr int DK      = 64;
static constexpr int SEQ     = 2048;
static constexpr int BATCH   = 2;
static constexpr int ROWS    = BATCH * SEQ;   // 4096
static constexpr int MAXLEN  = 512;

// Native fptrunc -> backend emits packed bf16 converts (v_cvt_pk_bf16_f32).
__device__ __forceinline__ __bf16 to_bf16(float f) { return (__bf16)f; }

// Concat two 8-wide bf16 vectors into a 16-wide WMMA fragment with no
// per-element moves (register-allocation-friendly).
__device__ __forceinline__ v16bf cat16(v8bf a, v8bf b) {
  return __builtin_shufflevector(a, b, 0, 1, 2, 3, 4, 5, 6, 7,
                                 8, 9, 10, 11, 12, 13, 14, 15);
}

// Convert 8 contiguous floats to v8bf.
__device__ __forceinline__ v8bf cvt8(const float* __restrict__ p) {
  v8bf r;
#pragma unroll
  for (int i = 0; i < 8; ++i) r[i] = (__bf16)p[i];
  return r;
}

// ---------------------------------------------------------------------------
// Projection GEMM: each wave computes one 16x16 tile of (X @ W^T).
// A-frag (16x32 bf16, ISA 7.12.2): lane M = l&15, k-base = (l>=16)*8,
//   elements {base..base+7} U {base+16..base+23}.
// B-frag (32x16 bf16): lane N = l&15, k-base = (l>=16)*8, same k order as A.
// blockIdx.z selects which projection (0=Q, 1=K, 2=V-transposed).
// ---------------------------------------------------------------------------
__global__ __launch_bounds__(128)
void proj_kernel(const float* __restrict__ query,
                 const float* __restrict__ key,
                 const float* __restrict__ Wq, const float* __restrict__ bq,
                 const float* __restrict__ Wk, const float* __restrict__ bk,
                 const float* __restrict__ Wv, const float* __restrict__ bv,
                 const float* __restrict__ pos,  // rel_pos_emb[MAXLEN-1], 64 floats
                 __bf16* __restrict__ Qb,        // [B,H,S,DK]
                 __bf16* __restrict__ Kb,        // [B,H,S,DK]
                 __bf16* __restrict__ Vt)        // [B,H,DK,S] (transposed)
{
  const int lane = threadIdx.x & 31;
  const int wave = threadIdx.x >> 5;
  const int lo   = lane & 15;
  const int hi   = lane >> 4;

  const int m0    = blockIdx.x * 16;               // x row tile
  const int n0    = blockIdx.y * 64 + wave * 16;   // output-feature tile
  const int which = blockIdx.z;

  const float* X    = (which == 0) ? query : key;
  const float* W    = (which == 0) ? Wq : (which == 1) ? Wk : Wv;
  const float* bias = (which == 0) ? bq : (which == 1) ? bk : bv;

  const int arow  = m0 + lo;
  const int abase = hi * 8;
  const int bcol  = n0 + lo;
  const int bkb   = hi * 8;

  const float* xrow = X + (size_t)arow * D_MODEL;
  const float* wrow = W + (size_t)bcol * D_MODEL;

  v8f acc = {};
  for (int k0 = 0; k0 < D_MODEL; k0 += 32) {
    __builtin_prefetch(xrow + k0 + 64, 0, 1);
    __builtin_prefetch(wrow + k0 + 64, 0, 1);
    const float* ap = xrow + k0 + abase;
    v16bf a = cat16(cvt8(ap), cvt8(ap + 16));
    const float* bp = wrow + k0 + bkb;
    v16bf bf = cat16(cvt8(bp), cvt8(bp + 16));
    acc = __builtin_amdgcn_wmma_f32_16x16x32_bf16(false, a, false, bf,
                                                  (short)0, acc, false, false);
  }

  // Epilogue: bias (+ zero-offset relative-position row for Q/K), 1/sqrt(DK)
  // folded into Q. C-layout: lane holds col N=bcol, rows r + 8*hi.
  const int h  = bcol >> 6;
  const int dk = bcol & 63;
  const float badd  = bias[bcol] + ((which <= 1) ? pos[dk] : 0.0f);
  const float scale = (which == 0) ? 0.125f : 1.0f;
  const int bidx = m0 >> 11;          // batch (SEQ = 2048)
  const int s0   = (m0 & (SEQ - 1));

  if (which == 2) {
    // transposed store: 8 consecutive s per lane -> one 16-byte packed store
    __bf16* dst = Vt + (((size_t)(bidx * HEADS + h) * DK + dk) * SEQ) + s0 + hi * 8;
    v8bf pk;
#pragma unroll
    for (int r = 0; r < 8; ++r) pk[r] = to_bf16(acc[r] + badd);
    *(v8bf*)dst = pk;
  } else {
    __bf16* dstb = (which == 0) ? Qb : Kb;
#pragma unroll
    for (int r = 0; r < 8; ++r) {
      const int s = s0 + r + hi * 8;
      dstb[(((size_t)(bidx * HEADS + h) * SEQ + s) * DK) + dk] =
          to_bf16((acc[r] + badd) * scale);
    }
  }
}

// ---------------------------------------------------------------------------
// Fused attention: each wave owns a 16-row Q tile of one (b,h); keys are
// consumed in 32-wide blocks with online softmax. 8 WMMAs per block.
// ---------------------------------------------------------------------------
__global__ __launch_bounds__(128)
void attn_kernel(const __bf16* __restrict__ Qb,
                 const __bf16* __restrict__ Kb,
                 const __bf16* __restrict__ Vt,
                 const int*    __restrict__ mask,
                 float*        __restrict__ out)
{
  __shared__ __align__(16) __bf16 lds_p[4][16][40];   // per-wave P tile, padded

  const int lane = threadIdx.x & 31;
  const int wave = threadIdx.x >> 5;
  const int lo   = lane & 15;
  const int hi   = lane >> 4;

  const int bh = blockIdx.y;            // b*HEADS + h
  const int b  = bh >> 4;
  const int h  = bh & 15;
  const int q0 = blockIdx.x * 64 + wave * 16;

  const __bf16* Qh = Qb + (size_t)bh * SEQ * DK;
  const __bf16* Kh = Kb + (size_t)bh * SEQ * DK;
  const __bf16* Vh = Vt + (size_t)bh * DK * SEQ;
  const int* mbase = mask + (size_t)b * SEQ * SEQ;

  // Persistent Q A-fragments (two 32-wide k-chunks of DK=64)
  v16bf qfrag[2];
  {
    const __bf16* qr = Qh + (size_t)(q0 + lo) * DK;
    const int abase = hi * 8;
#pragma unroll
    for (int dc = 0; dc < 2; ++dc) {
      qfrag[dc] = cat16(*(const v8bf*)(qr + dc * 32 + abase),
                        *(const v8bf*)(qr + dc * 32 + abase + 16));
    }
  }

  v8f o0 = {}, o1 = {}, o2 = {}, o3 = {};
  float rowmax[8], lsum[8];
#pragma unroll
  for (int r = 0; r < 8; ++r) { rowmax[r] = -1e30f; lsum[r] = 0.0f; }

  const int qsb = q0 + hi * 8;

  for (int kb = 0; kb < SEQ; kb += 32) {
    // --- scores: two 16x16 tiles, K accumulated over DK via 2 WMMAs each ---
    v8f c0 = {}, c1 = {};
#pragma unroll
    for (int t = 0; t < 2; ++t) {
      const __bf16* krow = Kh + (size_t)(kb + t * 16 + lo) * DK;
      v8f cacc = {};
#pragma unroll
      for (int dc = 0; dc < 2; ++dc) {
        const __bf16* p = krow + dc * 32 + hi * 8;
        v16bf bf = cat16(*(const v8bf*)p, *(const v8bf*)(p + 16));
        cacc = __builtin_amdgcn_wmma_f32_16x16x32_bf16(false, qfrag[dc], false, bf,
                                                       (short)0, cacc, false, false);
      }
      if (t == 0) c0 = cacc; else c1 = cacc;
    }

    // --- mask ---
#pragma unroll
    for (int r = 0; r < 8; ++r) {
      const int* mr = mbase + (size_t)(qsb + r) * SEQ + kb;
      if (mr[lo] == 0)      c0[r] = -1e9f;
      if (mr[16 + lo] == 0) c1[r] = -1e9f;
    }

    // --- online softmax (row reductions over 16 lanes of each half) ---
    float pr0[8], pr1[8];
#pragma unroll
    for (int r = 0; r < 8; ++r) {
      float mx = fmaxf(c0[r], c1[r]);
#pragma unroll
      for (int d = 8; d >= 1; d >>= 1) mx = fmaxf(mx, __shfl_xor(mx, d, 32));
      const float mn = fmaxf(rowmax[r], mx);
      const float sc = __expf(rowmax[r] - mn);
      rowmax[r] = mn;
      const float p0 = __expf(c0[r] - mn);
      const float p1 = __expf(c1[r] - mn);
      float sum = p0 + p1;
#pragma unroll
      for (int d = 8; d >= 1; d >>= 1) sum += __shfl_xor(sum, d, 32);
      lsum[r] = lsum[r] * sc + sum;
      o0[r] *= sc; o1[r] *= sc; o2[r] *= sc; o3[r] *= sc;
      pr0[r] = p0; pr1[r] = p1;
    }

    // --- C-layout -> A-layout for P via per-wave LDS tile (DS is in-order) ---
    __bf16 (*P)[40] = lds_p[wave];
#pragma unroll
    for (int r = 0; r < 8; ++r) {
      const int row = r + hi * 8;
      P[row][lo]      = to_bf16(pr0[r]);
      P[row][16 + lo] = to_bf16(pr1[r]);
    }
    v16bf pf = cat16(*(const v8bf*)&P[lo][hi * 8],
                     *(const v8bf*)&P[lo][hi * 8 + 16]);

    // --- O += P @ V : 4 output d-chunks, V^T rows give contiguous B-frags ---
    const int kvb = kb + hi * 8;
#pragma unroll
    for (int c = 0; c < 4; ++c) {
      const __bf16* vp = Vh + (size_t)(c * 16 + lo) * SEQ + kvb;
      v16bf vf = cat16(*(const v8bf*)vp, *(const v8bf*)(vp + 16));
      v8f* oc = (c == 0) ? &o0 : (c == 1) ? &o1 : (c == 2) ? &o2 : &o3;
      *oc = __builtin_amdgcn_wmma_f32_16x16x32_bf16(false, pf, false, vf,
                                                    (short)0, *oc, false, false);
    }
  }

  // --- finalize: divide by l, write out[b][s][h*64 + d] (fp32) ---
#pragma unroll
  for (int r = 0; r < 8; ++r) {
    const float inv = 1.0f / lsum[r];
    float* orow = out + ((size_t)b * SEQ + (qsb + r)) * (HEADS * DK) + h * DK;
    orow[ 0 + lo] = o0[r] * inv;
    orow[16 + lo] = o1[r] * inv;
    orow[32 + lo] = o2[r] * inv;
    orow[48 + lo] = o3[r] * inv;
  }
}

extern "C" void kernel_launch(void* const* d_in, const int* in_sizes, int n_in,
                              void* d_out, int out_size, void* d_ws, size_t ws_size,
                              hipStream_t stream) {
  const float* query = (const float*)d_in[0];
  const float* key   = (const float*)d_in[1];
  const int*   mask  = (const int*)  d_in[2];
  const float* Wq    = (const float*)d_in[3];
  const float* bq    = (const float*)d_in[4];
  const float* Wk    = (const float*)d_in[5];
  const float* bk    = (const float*)d_in[6];
  const float* Wv    = (const float*)d_in[7];
  const float* bv    = (const float*)d_in[8];
  const float* rel   = (const float*)d_in[9];
  const float* pos   = rel + (size_t)(MAXLEN - 1) * DK;  // zero-offset row

  // Workspace: Q, K (bf16 [B,H,S,DK]) + V^T (bf16 [B,H,DK,S]) = 24 MB total.
  const size_t per = (size_t)BATCH * HEADS * SEQ * DK;
  __bf16* Qb = (__bf16*)d_ws;
  __bf16* Kb = Qb + per;
  __bf16* Vt = Kb + per;

  dim3 pg(ROWS / 16, D_MODEL / 64, 3);   // 256 x 16 x 3 blocks, 128 thr (4 waves)
  proj_kernel<<<pg, dim3(128), 0, stream>>>(query, key, Wq, bq, Wk, bk, Wv, bv,
                                            pos, Qb, Kb, Vt);

  dim3 ag(SEQ / 64, BATCH * HEADS);      // 32 x 32 blocks, 128 thr (4 waves)
  attn_kernel<<<ag, dim3(128), 0, stream>>>(Qb, Kb, Vt, mask, (float*)d_out);
}
